// MyModel_83537113907498
// MI455X (gfx1250) — matrix operand order repaired and checked by reading, not compiled, not run on hardware
//
#include <hip/hip_runtime.h>
#include <hip/hip_bf16.h>

typedef __attribute__((ext_vector_type(16))) __bf16 v16bf;
typedef __attribute__((ext_vector_type(2)))  __bf16 v2bf;
typedef __attribute__((ext_vector_type(8)))  float  v8f;
typedef __attribute__((ext_vector_type(4)))  unsigned int uint4v;

union Frag {
  v16bf  bf;
  uint4v u[2];
};

#define HQ    32
#define HKV   8
#define SQ    4096
#define SKV   8192
#define DIM   128
#define KVBLK 64
#define NSEL  16
#define NQB   64
#define QBLK  64
#define GROUP 4
#define NTHR  512   // 16 waves: 4 group heads x 4 waves per head

// ---- fp32 -> bf16 (RNE) ----
__device__ __forceinline__ unsigned int f2bf(float f) {
  unsigned int u = __float_as_uint(f);
  return (u + 0x7FFFu + ((u >> 16) & 1u)) >> 16;
}

#if __has_builtin(__builtin_amdgcn_cvt_pk_bf16_f32)
__device__ __forceinline__ unsigned int pk2bf(float a, float b) {
  union { v2bf v; unsigned int u; } cv;
  cv.v = __builtin_amdgcn_cvt_pk_bf16_f32(a, b);   // lo=a, hi=b
  return cv.u;
}
#else
__device__ __forceinline__ unsigned int pk2bf(float a, float b) {
  return f2bf(a) | (f2bf(b) << 16);
}
#endif

#if __has_builtin(__builtin_amdgcn_exp2f)
#define EXP2F(x) __builtin_amdgcn_exp2f(x)
#else
#define EXP2F(x) exp2f(x)
#endif

// ---- K prefetch: 4 x b128 per thread ----
__device__ __forceinline__ void load_k_regs(const float* __restrict__ kbase,
                                            int tid, float4 kr[4]) {
  const float4* k4 = (const float4*)kbase;
#pragma unroll
  for (int j = 0; j < 4; ++j) kr[j] = k4[tid + j * NTHR];
}

__device__ __forceinline__ void store_k_lds(unsigned short* kdst, int tid,
                                            const float4 kr[4]) {
  unsigned int* ks32 = (unsigned int*)kdst;
#pragma unroll
  for (int j = 0; j < 4; ++j) {
    int i = tid + j * NTHR;
    ks32[i * 2 + 0] = pk2bf(kr[j].x, kr[j].y);
    ks32[i * 2 + 1] = pk2bf(kr[j].z, kr[j].w);
  }
}

// ---- V prefetch: 4 x b128 per thread (row pairs, for transposed store) ----
__device__ __forceinline__ void load_v_regs(const float* __restrict__ vbase,
                                            int tid, float4 vr[2][2]) {
  const float4* v4 = (const float4*)vbase;
#pragma unroll
  for (int j = 0; j < 2; ++j) {
    int i  = tid + j * NTHR;
    int rp = i >> 5, c4 = i & 31;            // kv row-pair, float4-in-row
    vr[j][0] = v4[(2 * rp + 0) * (DIM / 4) + c4];
    vr[j][1] = v4[(2 * rp + 1) * (DIM / 4) + c4];
  }
}

__device__ __forceinline__ void store_v_lds(unsigned short* vdst, int tid,
                                            const float4 vr[2][2]) {
  unsigned int* vt32 = (unsigned int*)vdst;
#pragma unroll
  for (int j = 0; j < 2; ++j) {
    int i  = tid + j * NTHR;
    int rp = i >> 5, c4 = i & 31;
    vt32[(c4 * 4 + 0) * (KVBLK / 2) + rp] = pk2bf(vr[j][0].x, vr[j][1].x);
    vt32[(c4 * 4 + 1) * (KVBLK / 2) + rp] = pk2bf(vr[j][0].y, vr[j][1].y);
    vt32[(c4 * 4 + 2) * (KVBLK / 2) + rp] = pk2bf(vr[j][0].z, vr[j][1].z);
    vt32[(c4 * 4 + 3) * (KVBLK / 2) + rp] = pk2bf(vr[j][0].w, vr[j][1].w);
  }
}

__global__ __launch_bounds__(NTHR, 1) void nsa_sparse_attn_kernel(
    const float* __restrict__ q, const float* __restrict__ k,
    const float* __restrict__ v, const long long* __restrict__ idx,
    float* __restrict__ out) {
  const int qb   = blockIdx.x;        // query block 0..63
  const int hkv  = blockIdx.y;        // kv head    0..7
  const int tid  = threadIdx.x;
  const int wave = tid >> 5;          // 0..15
  const int g    = wave >> 2;         // head within GQA group 0..3
  const int wh   = wave & 3;          // wave within head (16 query rows each)
  const int lane = tid & 31;
  const int ln   = lane & 15;
  const int half = lane >> 4;
  const float scale = 0.088388347648318447f;          // 1/sqrt(128)
  const float c     = scale * 1.4426950408889634f;    // scale * log2(e)

  // double-buffered K/V + per-wave P staging: 32+32+32 = 96 KB LDS
  __shared__ __align__(16) unsigned short k_s[2][KVBLK * DIM];
  __shared__ __align__(16) unsigned short vT_s[2][DIM * KVBLK];
  __shared__ __align__(16) unsigned short p_s[16][16 * KVBLK];

  // ---- Q A-fragments straight from global (each wave reads its own 16 rows once) ----
  // A 16-bit layout: lane row m=ln; element e -> d = kc*32 + 16*(e/8) + half*8 + e%8
  Frag qf[4];
  {
    const float* qrow = q + ((size_t)(hkv * GROUP + g) * SQ +
                             (size_t)qb * QBLK + wh * 16 + ln) * DIM;
#pragma unroll
    for (int kc = 0; kc < 4; ++kc) {
      const float4* pA = (const float4*)(qrow + kc * 32 + half * 8);
      const float4* pB = (const float4*)(qrow + kc * 32 + 16 + half * 8);
      float4 a0 = pA[0], a1 = pA[1];
      float4 b0 = pB[0], b1 = pB[1];
      uint4v u0, u1;
      u0.x = pk2bf(a0.x, a0.y); u0.y = pk2bf(a0.z, a0.w);
      u0.z = pk2bf(a1.x, a1.y); u0.w = pk2bf(a1.z, a1.w);
      u1.x = pk2bf(b0.x, b0.y); u1.y = pk2bf(b0.z, b0.w);
      u1.z = pk2bf(b1.x, b1.y); u1.w = pk2bf(b1.z, b1.w);
      qf[kc].u[0] = u0;
      qf[kc].u[1] = u1;
    }
  }

  float m_i[8], l_i[8];
  v8f oacc[8];
#pragma unroll
  for (int r = 0; r < 8; ++r) { m_i[r] = -__builtin_inff(); l_i[r] = 0.0f; }
#pragma unroll
  for (int ot = 0; ot < 8; ++ot)
#pragma unroll
    for (int r = 0; r < 8; ++r) oacc[ot][r] = 0.0f;

  const long long* idxbase = idx + ((size_t)hkv * NQB + qb) * NSEL;
  const float* khead = k + (size_t)hkv * SKV * DIM;
  const float* vhead = v + (size_t)hkv * SKV * DIM;

  // ---- prologue: stage block 0 into LDS buffer 0 ----
  {
    float4 kr0[4], vr0[2][2];
    const int blk0 = (int)idxbase[0];
    load_k_regs(khead + (size_t)blk0 * KVBLK * DIM, tid, kr0);
    load_v_regs(vhead + (size_t)blk0 * KVBLK * DIM, tid, vr0);
    store_k_lds(k_s[0], tid, kr0);
    store_v_lds(vT_s[0], tid, vr0);
  }
  __syncthreads();

#pragma unroll 2
  for (int s = 0; s < NSEL; ++s) {
    const int cur = s & 1;                    // compile-time under unroll-2
    const unsigned short* ks = k_s[cur];
    const unsigned short* vt = vT_s[cur];
    const bool havenext = (s + 1 < NSEL);
    const int blkn = havenext ? (int)idxbase[s + 1] : 0;

    // issue next K block's global loads; they land during S + softmax
    float4 kr[4];
    if (havenext)
      load_k_regs(khead + (size_t)blkn * KVBLK * DIM, tid, kr);

    // ---- S = Q @ K^T : 16 rows x 64 kv cols per wave ----
    v8f sacc[4];
#pragma unroll
    for (int nt = 0; nt < 4; ++nt)
#pragma unroll
      for (int r = 0; r < 8; ++r) sacc[nt][r] = 0.0f;

#pragma unroll
    for (int kc = 0; kc < 4; ++kc) {
#pragma unroll
      for (int nt = 0; nt < 4; ++nt) {
        // B 16-bit layout: lane col n=ln; element e -> K(=d) = kc*32 + 16*(e/8) + half*8 + e%8
        Frag kb;
        int base = (nt * 16 + ln) * DIM + kc * 32 + half * 8;
        kb.u[0] = *(const uint4v*)&ks[base];
        kb.u[1] = *(const uint4v*)&ks[base + 16];
        sacc[nt] = __builtin_amdgcn_wmma_f32_16x16x32_bf16(
            false, qf[kc].bf, false, kb.bf, (short)0, sacc[nt], false, false);
      }
    }

    // ---- online softmax over the 64 selected columns ----
    // C layout: sacc[nt][r] = S[row = r + 8*half][col = nt*16 + ln]
    float corr[8];
#pragma unroll
    for (int r = 0; r < 8; ++r) {
      float mx = fmaxf(fmaxf(sacc[0][r], sacc[1][r]), fmaxf(sacc[2][r], sacc[3][r]));
#pragma unroll
      for (int msk = 1; msk < 16; msk <<= 1)
        mx = fmaxf(mx, __shfl_xor(mx, msk, 32));
      float mnew  = fmaxf(m_i[r], mx);
      float mnewc = mnew * c;
      corr[r] = EXP2F(m_i[r] * c - mnewc);
      float psum = 0.0f;
      int prow = (r + 8 * half) * KVBLK + ln;
#pragma unroll
      for (int nt = 0; nt < 4; ++nt) {
        float p = EXP2F(fmaf(sacc[nt][r], c, -mnewc));
        psum += p;
        p_s[wave][prow + nt * 16] = (unsigned short)f2bf(p);
      }
#pragma unroll
      for (int msk = 1; msk < 16; msk <<= 1)
        psum += __shfl_xor(psum, msk, 32);
      l_i[r] = l_i[r] * corr[r] + psum;
      m_i[r] = mnew;
    }
#pragma unroll
    for (int ot = 0; ot < 8; ++ot)
#pragma unroll
      for (int r = 0; r < 8; ++r) oacc[ot][r] *= corr[r];

    // retire K prefetch (loads had the whole S+softmax window), free its regs
    if (havenext)
      store_k_lds(k_s[cur ^ 1], tid, kr);

    // issue next V block's global loads; they land during P@V
    float4 vr[2][2];
    if (havenext)
      load_v_regs(vhead + (size_t)blkn * KVBLK * DIM, tid, vr);

    // ---- O += P @ V ----
    Frag pf[2];
#pragma unroll
    for (int kc = 0; kc < 2; ++kc) {
      int base = ln * KVBLK + kc * 32 + half * 8;
      pf[kc].u[0] = *(const uint4v*)&p_s[wave][base];
      pf[kc].u[1] = *(const uint4v*)&p_s[wave][base + 16];
    }
#pragma unroll
    for (int ot = 0; ot < 8; ++ot) {
#pragma unroll
      for (int kc = 0; kc < 2; ++kc) {
        // B element e -> K(=kv) = kc*32 + 16*(e/8) + half*8 + e%8 ; col n = d = ot*16 + ln
        Frag vb;
        int base = (ot * 16 + ln) * KVBLK + kc * 32 + half * 8;
        vb.u[0] = *(const uint4v*)&vt[base];
        vb.u[1] = *(const uint4v*)&vt[base + 16];
        oacc[ot] = __builtin_amdgcn_wmma_f32_16x16x32_bf16(
            false, pf[kc].bf, false, vb.bf, (short)0, oacc[ot], false, false);
      }
    }

    // retire V prefetch into the other LDS buffer
    if (havenext)
      store_v_lds(vT_s[cur ^ 1], tid, vr);
    __syncthreads();
  }

  // ---- epilogue: normalize by softmax denominator, store fp32 ----
  {
    float* obase = out + ((size_t)(hkv * GROUP + g) * SQ +
                          (size_t)qb * QBLK + (size_t)wh * 16) * DIM;
#pragma unroll
    for (int r = 0; r < 8; ++r) {
      float inv = 1.0f / l_i[r];
      int row = r + 8 * half;
#pragma unroll
      for (int ot = 0; ot < 8; ++ot)
        obase[(size_t)row * DIM + ot * 16 + ln] = oacc[ot][r] * inv;
    }
  }
}

extern "C" void kernel_launch(void* const* d_in, const int* in_sizes, int n_in,
                              void* d_out, int out_size, void* d_ws, size_t ws_size,
                              hipStream_t stream) {
  const float*     q   = (const float*)d_in[0];
  const float*     k   = (const float*)d_in[1];
  const float*     v   = (const float*)d_in[2];
  const long long* idx = (const long long*)d_in[3];
  float*           out = (float*)d_out;
  (void)in_sizes; (void)n_in; (void)out_size; (void)d_ws; (void)ws_size;

  dim3 grid(NQB, HKV);   // 64 query blocks x 8 kv heads (GQA group fused per block)
  nsa_sparse_attn_kernel<<<grid, NTHR, 0, stream>>>(q, k, v, idx, out);
}
